// AMLDetector_60988535603848
// MI455X (gfx1250) — compile-verified, matrix operand order checked by reading, not run on hardware
//
#include <hip/hip_runtime.h>
#include <cstdint>
#include <cstddef>

// ---------------------------------------------------------------------------
// AMLDetector GNN forward for MI455X (gfx1250), wave32 + WMMA f16 path.
// All dense GEMMs (~1.05 TFLOP total, dominated by the JK-LSTM chain) run on
// v_wmma_f32_16x16x32_f16. A and B tiles are staged in LDS so each fragment is
// two contiguous 16B chunks -> ds_load_b128. Interior blocks use vectorized
// (v8h) staging; only edge tiles take the guarded scalar path.
// Edge softmax/scatter uses f32 global atomics. Workspace: ~1.65 GB in d_ws.
// ---------------------------------------------------------------------------

#define NHEADS 4
#define CCH    128
#define DD     512
#define EDIM   32
#define JKH    768
#define G4     3072   // 4*JKH

typedef __attribute__((ext_vector_type(16))) _Float16 v16h;
typedef __attribute__((ext_vector_type(8)))  _Float16 v8h;
typedef __attribute__((ext_vector_type(8)))  float    v8f;

__device__ __forceinline__ float elu1(float x){ return x > 0.f ? x : (__expf(x) - 1.f); }
__device__ __forceinline__ float sigm(float x){ return 1.f / (1.f + __expf(-x)); }

__device__ __forceinline__ float wave_sum(float v){
#pragma unroll
  for (int o = 16; o; o >>= 1) v += __shfl_xor(v, o, 32);
  return v;
}

__device__ void atomicMaxF(float* addr, float val){
  unsigned int* ai = reinterpret_cast<unsigned int*>(addr);
  unsigned int cur = *ai;
  while (__uint_as_float(cur) < val) {
    unsigned int prev = atomicCAS(ai, cur, __float_as_uint(val));
    if (prev == cur) break;
    cur = prev;
  }
}

// Load one 16x32 f16 fragment row: halves [kb..kb+7] and [16+kb..23+kb] of a
// 16B-aligned LDS row -> two ds_load_b128, merged into a v16h.
__device__ __forceinline__ v16h frag_ld(const _Float16* rowbase, int kb){
  v8h lo = *(const v8h*)(rowbase + kb);
  v8h hi = *(const v8h*)(rowbase + 16 + kb);
  return __builtin_shufflevector(lo, hi, 0,1,2,3,4,5,6,7,8,9,10,11,12,13,14,15);
}

// ---------------------------------------------------------------------------
// Utility kernels
// ---------------------------------------------------------------------------
__global__ void k_fill(float* p, float v, size_t n){
  size_t i = (size_t)blockIdx.x * blockDim.x + threadIdx.x;
  if (i < n) p[i] = v;
}

__global__ void k_cvt16(const float* __restrict__ s, _Float16* __restrict__ d, size_t n){
  size_t i = (size_t)blockIdx.x * blockDim.x + threadIdx.x;
  if (i < n) d[i] = (_Float16)s[i];
}

// src [R,C] row-major -> dst [C,R] row-major (for W used as x @ W^T)
__global__ void k_cvt16_tr(const float* __restrict__ s, _Float16* __restrict__ d, int R, int C){
  size_t i = (size_t)blockIdx.x * blockDim.x + threadIdx.x;
  if (i < (size_t)R * C){
    int r = (int)(i / C), c = (int)(i % C);
    d[(size_t)c * R + r] = (_Float16)s[i];
  }
}

// ---------------------------------------------------------------------------
// Generic WMMA GEMM: C[M,N] = A[M,K](f16) @ B[K,N](f16)  (+ bias/ELU/acc/f16out)
// Block: 256 threads = 8 waves; block tile 128x64; wave tile 32x32 (2x2 WMMA).
// flags: 1=add bias[N], 2=ELU, 4=accumulate into existing f32 C, 8=store f16
// ---------------------------------------------------------------------------
__global__ __launch_bounds__(256)
void k_gemm_wmma(const _Float16* __restrict__ A, const _Float16* __restrict__ B,
                 const float* __restrict__ bias, float* __restrict__ Cout,
                 int M, int N, int K, int flags)
{
  constexpr int BM = 128, BN = 64, BK = 32;
  __shared__ _Float16 As [BM][BK + 8];   // row-major A tile (row stride 80B)
  __shared__ _Float16 Bst[BN][BK + 8];   // TRANSPOSED B tile: [n][k]

  const int tid  = threadIdx.x;
  const int lane = tid & 31;
  const int wv   = tid >> 5;
  const int wm   = (wv >> 1) * 32;   // 4 waves along M
  const int wn   = (wv & 1) * 32;    // 2 waves along N
  const int bm0  = blockIdx.x * BM;
  const int bn0  = blockIdx.y * BN;

  // Uniform fast-path predicate: whole tile in bounds, rows 16B-aligned.
  const bool interior = (bm0 + BM <= M) && (bn0 + BN <= N) &&
                        ((K & (BK - 1)) == 0) && ((N & 7) == 0);

  v8f acc[2][2] = {};

  for (int k0 = 0; k0 < K; k0 += BK) {
    // prefetch next K tiles into cache (global_prefetch_b8)
    if (k0 + BK < K) {
      int pr = bm0 + (tid >> 1);
      if (pr < M) __builtin_prefetch(&A[(size_t)pr * K + k0 + BK + (tid & 1) * 16], 0, 3);
      int pk = k0 + BK + (tid >> 5);
      int pn = bn0 + (tid & 31) * 2;
      if (pk < K && pn < N) __builtin_prefetch(&B[(size_t)pk * N + pn], 0, 3);
    }
    if (interior) {
      // A: two 8-half (16B) chunks per thread -> global b128 + ds_store_b128
      {
        int q0 = tid, q1 = tid + 256;
        int r0 = q0 >> 2, c0 = (q0 & 3) * 8;
        int r1 = q1 >> 2, c1 = (q1 & 3) * 8;
        *(v8h*)&As[r0][c0] = *(const v8h*)&A[(size_t)(bm0 + r0) * K + k0 + c0];
        *(v8h*)&As[r1][c1] = *(const v8h*)&A[(size_t)(bm0 + r1) * K + k0 + c1];
      }
      // B: one coalesced 8-half chunk per thread, transposed into Bst[n][k]
      {
        int kk = tid >> 3;           // 0..31
        int n0 = (tid & 7) * 8;      // 0..56
        v8h bv = *(const v8h*)&B[(size_t)(k0 + kk) * N + bn0 + n0];
#pragma unroll
        for (int j = 0; j < 8; j++) Bst[n0 + j][kk] = bv[j];
      }
    } else {
      // guarded scalar staging (edge tiles only)
      for (int idx = tid; idx < BM * BK; idx += 256) {
        int r = idx >> 5, c = idx & 31;
        int gm = bm0 + r, gk = k0 + c;
        As[r][c] = (gm < M && gk < K) ? A[(size_t)gm * K + gk] : (_Float16)0.f;
      }
      for (int idx = tid; idx < BK * BN; idx += 256) {
        int r = idx >> 6, c = idx & 63;          // r = k, c = n
        int gk = k0 + r, gn = bn0 + c;
        Bst[c][r] = (gk < K && gn < N) ? B[(size_t)gk * N + gn] : (_Float16)0.f;
      }
    }
    __syncthreads();

    // 16-bit 16x32 fragment layout (ISA 7.12.2): lanes 0-15 hold K 0..7 &
    // 16..23, lanes 16-31 hold K 8..15 & 24..31. Same shape for A (M across
    // lanes) and B (N across lanes, since Bst is N-major).
    const int kb = (lane >> 4) * 8;
    const int mr = lane & 15;
    v16h afrag[2], bfrag[2];
#pragma unroll
    for (int mt = 0; mt < 2; mt++)
      afrag[mt] = frag_ld(&As[wm + mt * 16 + mr][0], kb);
#pragma unroll
    for (int nt = 0; nt < 2; nt++)
      bfrag[nt] = frag_ld(&Bst[wn + nt * 16 + mr][0], kb);

#pragma unroll
    for (int mt = 0; mt < 2; mt++)
#pragma unroll
      for (int nt = 0; nt < 2; nt++)
        acc[mt][nt] = __builtin_amdgcn_wmma_f32_16x16x32_f16(
            false, afrag[mt], false, bfrag[nt], (short)0, acc[mt][nt], false, false);
    __syncthreads();
  }

  // C/D layout: VGPR r -> M = r (+8 for lanes 16-31), N = lane&15
  _Float16* C16 = (_Float16*)Cout;
#pragma unroll
  for (int mt = 0; mt < 2; mt++)
#pragma unroll
    for (int nt = 0; nt < 2; nt++){
      int col = bn0 + wn + nt * 16 + (lane & 15);
      if (col >= N) continue;
      int rbase = bm0 + wm + mt * 16 + ((lane >> 4) << 3);
#pragma unroll
      for (int r = 0; r < 8; r++){
        int row = rbase + r;
        if (row >= M) continue;
        float v = acc[mt][nt][r];
        if (flags & 1) v += bias[col];
        if (flags & 4) v += Cout[(size_t)row * N + col];
        if (flags & 2) v = elu1(v);
        if (flags & 8) C16[(size_t)row * N + col] = (_Float16)v;
        else           Cout[(size_t)row * N + col] = v;
      }
    }
}

// ---------------------------------------------------------------------------
// Input projection LayerNorm(256) + ELU, in-place. One block (256 thr) / row.
// ---------------------------------------------------------------------------
__global__ __launch_bounds__(256)
void k_ln_elu(float* __restrict__ x, const float* __restrict__ g, const float* __restrict__ b){
  const int row = blockIdx.x, tid = threadIdx.x;
  __shared__ float part[8];
  __shared__ float stat;
  float v = x[(size_t)row * 256 + tid];
  float s = wave_sum(v);
  if ((tid & 31) == 0) part[tid >> 5] = s;
  __syncthreads();
  if (tid == 0){ float t = 0.f; for (int i = 0; i < 8; i++) t += part[i]; stat = t / 256.f; }
  __syncthreads();
  const float mu = stat;
  const float d  = v - mu;
  __syncthreads();
  s = wave_sum(d * d);
  if ((tid & 31) == 0) part[tid >> 5] = s;
  __syncthreads();
  if (tid == 0){ float t = 0.f; for (int i = 0; i < 8; i++) t += part[i]; stat = t / 256.f; }
  __syncthreads();
  float o = d * rsqrtf(stat + 1e-5f) * g[tid] + b[tid];
  x[(size_t)row * 256 + tid] = elu1(o);
}

// ---------------------------------------------------------------------------
// Edge embedding: e = elu(ea @ W1[3,32] + b1) @ W2[32,32] + b2   (tiny K)
// ---------------------------------------------------------------------------
__global__ void k_edge_embed(const float* __restrict__ ea,
                             const float* __restrict__ W1, const float* __restrict__ b1,
                             const float* __restrict__ W2, const float* __restrict__ b2,
                             float* __restrict__ eo, int E){
  int e = blockIdx.x * blockDim.x + threadIdx.x;
  if (e >= E) return;
  float a0 = ea[e * 3], a1 = ea[e * 3 + 1], a2 = ea[e * 3 + 2];
  float h[32];
#pragma unroll
  for (int j = 0; j < 32; j++)
    h[j] = elu1(a0 * W1[j] + a1 * W1[32 + j] + a2 * W1[64 + j] + b1[j]);
#pragma unroll
  for (int j2 = 0; j2 < 32; j2++){
    float s = b2[j2];
#pragma unroll
    for (int j = 0; j < 32; j++) s += h[j] * W2[j * 32 + j2];
    eo[(size_t)e * 32 + j2] = s;
  }
}

// ---------------------------------------------------------------------------
// GATv2 edge logits: wave per (edge,head); dot(leakyrelu(xl[src]+xr[dst]+ea), att)
// ---------------------------------------------------------------------------
__global__ void k_gat_logits(const float* __restrict__ xl, const float* __restrict__ xr,
                             const _Float16* __restrict__ eaf, const int* __restrict__ src,
                             const int* __restrict__ dst, const float* __restrict__ att,
                             float* __restrict__ logits, float* __restrict__ segmax, int E){
  int gw = blockIdx.x * 8 + (threadIdx.x >> 5);
  if (gw >= E * NHEADS) return;
  int e = gw >> 2, h = gw & 3, lane = threadIdx.x & 31;
  int s = src[e], d = dst[e];
  const float*    pl = xl  + (size_t)s * DD + h * CCH;
  const float*    pr = xr  + (size_t)d * DD + h * CCH;
  const _Float16* pe = eaf + (size_t)e * DD + h * CCH;
  const float*    pa = att + h * CCH;
  float acc = 0.f;
  for (int c = lane; c < CCH; c += 32){
    float z = pl[c] + pr[c] + (float)pe[c];
    z = z > 0.f ? z : 0.2f * z;
    acc += z * pa[c];
  }
  acc = wave_sum(acc);
  if (lane == 0){
    logits[(size_t)e * NHEADS + h] = acc;
    atomicMaxF(&segmax[(size_t)d * NHEADS + h], acc);
  }
}

// exp(logit - segmax[dst]) ; accumulate segment sums (shared by GAT & Transformer)
__global__ void k_seg_expsum(float* __restrict__ logits, const int* __restrict__ dst,
                             const float* __restrict__ segmax, float* __restrict__ segsum, int E){
  int i = blockIdx.x * blockDim.x + threadIdx.x;
  if (i >= E * NHEADS) return;
  int e = i >> 2, h = i & 3;
  int d = dst[e];
  float v = __expf(logits[i] - segmax[(size_t)d * NHEADS + h]);
  logits[i] = v;
  atomicAdd(&segsum[(size_t)d * NHEADS + h], v);
}

// GATv2 scatter: out[dst] += xl[src] * alpha
__global__ void k_gat_scatter(const float* __restrict__ xl, const float* __restrict__ eexp,
                              const float* __restrict__ segsum, const int* __restrict__ src,
                              const int* __restrict__ dst, float* __restrict__ out, int E){
  int gw = blockIdx.x * 8 + (threadIdx.x >> 5);
  if (gw >= E * NHEADS) return;
  int e = gw >> 2, h = gw & 3, lane = threadIdx.x & 31;
  int s = src[e], d = dst[e];
  float alpha = eexp[(size_t)e * NHEADS + h] / (segsum[(size_t)d * NHEADS + h] + 1e-16f);
  const float* pl = xl + (size_t)s * DD + h * CCH;
  float* po = out + (size_t)d * DD + h * CCH;
  for (int c = lane; c < CCH; c += 32)
    atomicAdd(&po[c], pl[c] * alpha);
}

// TransformerConv logits: dot(q[dst], k[src]+ea) / sqrt(C)
__global__ void k_trans_logits(const float* __restrict__ q, const float* __restrict__ k,
                               const _Float16* __restrict__ eaf, const int* __restrict__ src,
                               const int* __restrict__ dst, float* __restrict__ logits,
                               float* __restrict__ segmax, int E){
  int gw = blockIdx.x * 8 + (threadIdx.x >> 5);
  if (gw >= E * NHEADS) return;
  int e = gw >> 2, h = gw & 3, lane = threadIdx.x & 31;
  int s = src[e], d = dst[e];
  const float*    pq = q   + (size_t)d * DD + h * CCH;
  const float*    pk = k   + (size_t)s * DD + h * CCH;
  const _Float16* pe = eaf + (size_t)e * DD + h * CCH;
  float acc = 0.f;
  for (int c = lane; c < CCH; c += 32)
    acc += pq[c] * (pk[c] + (float)pe[c]);
  acc = wave_sum(acc) * 0.08838834764831845f;   // 1/sqrt(128)
  if (lane == 0){
    logits[(size_t)e * NHEADS + h] = acc;
    atomicMaxF(&segmax[(size_t)d * NHEADS + h], acc);
  }
}

// Transformer scatter: out[dst] += (v[src]+ea) * alpha
__global__ void k_trans_scatter(const float* __restrict__ v, const _Float16* __restrict__ eaf,
                                const float* __restrict__ eexp, const float* __restrict__ segsum,
                                const int* __restrict__ src, const int* __restrict__ dst,
                                float* __restrict__ out, int E){
  int gw = blockIdx.x * 8 + (threadIdx.x >> 5);
  if (gw >= E * NHEADS) return;
  int e = gw >> 2, h = gw & 3, lane = threadIdx.x & 31;
  int s = src[e], d = dst[e];
  float alpha = eexp[(size_t)e * NHEADS + h] / (segsum[(size_t)d * NHEADS + h] + 1e-16f);
  const float*    pv = v   + (size_t)s * DD + h * CCH;
  const _Float16* pe = eaf + (size_t)e * DD + h * CCH;
  float* po = out + (size_t)d * DD + h * CCH;
  for (int c = lane; c < CCH; c += 32)
    atomicAdd(&po[c], (pv[c] + (float)pe[c]) * alpha);
}

// out = elu(bn(in + cbias)) (+ residual)
__global__ void k_bn_elu(const float* __restrict__ in, const float* __restrict__ cbias,
                         const float* __restrict__ g, const float* __restrict__ b,
                         const float* __restrict__ rm, const float* __restrict__ rv,
                         const float* __restrict__ res, float* __restrict__ out, size_t total){
  size_t i = (size_t)blockIdx.x * blockDim.x + threadIdx.x;
  if (i >= total) return;
  int c = (int)(i % DD);
  float v = in[i] + (cbias ? cbias[c] : 0.f);
  v = (v - rm[c]) * rsqrtf(rv[c] + 1e-5f) * g[c] + b[c];
  v = elu1(v);
  if (res) v += res[i];
  out[i] = v;
}

// beta gate: res = beta*x_r + (1-beta)*out, beta = sigmoid([out,x_r,out-x_r]@Wb)
__global__ void k_beta(const float* __restrict__ outb, const float* __restrict__ xskip,
                       const float* __restrict__ Wb, float* __restrict__ res, int n_nodes){
  int gw = blockIdx.x * 8 + (threadIdx.x >> 5);
  if (gw >= n_nodes) return;
  int lane = threadIdx.x & 31;
  const float* po = outb  + (size_t)gw * DD;
  const float* px = xskip + (size_t)gw * DD;
  float s = 0.f;
  for (int i = lane; i < DD; i += 32){
    float o = po[i], x = px[i];
    s += o * Wb[i] + x * Wb[DD + i] + (o - x) * Wb[2 * DD + i];
  }
  s = wave_sum(s);
  float beta = sigm(s);
  float* pr = res + (size_t)gw * DD;
  for (int i = lane; i < DD; i += 32)
    pr[i] = beta * px[i] + (1.f - beta) * po[i];
}

// LSTM cell (torch gate order i,f,g,o); G = x@Wih^T + h@Whh^T (no biases yet)
__global__ void k_lstm_cell(const float* __restrict__ G, const float* __restrict__ bih,
                            const float* __restrict__ bhh, float* __restrict__ h,
                            float* __restrict__ c, float* __restrict__ hs, int n_nodes){
  size_t i = (size_t)blockIdx.x * blockDim.x + threadIdx.x;
  if (i >= (size_t)n_nodes * JKH) return;
  size_t n = i / JKH;
  int j = (int)(i % JKH);
  const float* Gp = G + n * (size_t)G4;
  float ig = Gp[j]           + bih[j]           + bhh[j];
  float fg = Gp[JKH + j]     + bih[JKH + j]     + bhh[JKH + j];
  float gg = Gp[2 * JKH + j] + bih[2 * JKH + j] + bhh[2 * JKH + j];
  float og = Gp[3 * JKH + j] + bih[3 * JKH + j] + bhh[3 * JKH + j];
  float cc = sigm(fg) * c[i] + sigm(ig) * tanhf(gg);
  float hh = sigm(og) * tanhf(cc);
  c[i] = cc; h[i] = hh; hs[i] = hh;
}

// JK attention over layers: a_l = [hf_l, hb_l] @ att_W + att_b; softmax over l
__global__ void k_jk_attn(const float* __restrict__ hf, const float* __restrict__ hb,
                          const float* __restrict__ attW, const float* __restrict__ attb,
                          const float* __restrict__ l1, const float* __restrict__ l2,
                          const float* __restrict__ l3, float* __restrict__ jk, int n_nodes){
  int gw = blockIdx.x * 8 + (threadIdx.x >> 5);
  if (gw >= n_nodes) return;
  int lane = threadIdx.x & 31;
  float a[3];
#pragma unroll
  for (int l = 0; l < 3; l++){
    const float* pf = hf + ((size_t)l * n_nodes + gw) * JKH;
    const float* pb = hb + ((size_t)l * n_nodes + gw) * JKH;
    float s = 0.f;
    for (int i = lane; i < JKH; i += 32)
      s += pf[i] * attW[i] + pb[i] * attW[JKH + i];
    a[l] = wave_sum(s) + attb[0];
  }
  float m = fmaxf(a[0], fmaxf(a[1], a[2]));
  float e0 = __expf(a[0] - m), e1 = __expf(a[1] - m), e2 = __expf(a[2] - m);
  float inv = 1.f / (e0 + e1 + e2);
  e0 *= inv; e1 *= inv; e2 *= inv;
  float* pj = jk + (size_t)gw * DD;
  for (int i = lane; i < DD; i += 32)
    pj[i] = e0 * l1[(size_t)gw * DD + i] + e1 * l2[(size_t)gw * DD + i] + e2 * l3[(size_t)gw * DD + i];
}

// ---------------------------------------------------------------------------
// Input index map: setup_inputs() dict flattened in insertion order.
// ---------------------------------------------------------------------------
enum {
  IN_X = 0, IN_EI, IN_EA,
  P_IN_W, P_IN_B, P_LN_G, P_LN_B,
  P_EM_W1, P_EM_B1, P_EM_W2, P_EM_B2,
  C1_WL, C1_BL, C1_WR, C1_BR, C1_WE, C1_ATT, C1_BIAS,
  BN1_G, BN1_B, BN1_RM, BN1_RV,
  C2_WL, C2_BL, C2_WR, C2_BR, C2_WE, C2_ATT, C2_BIAS,
  BN2_G, BN2_B, BN2_RM, BN2_RV,
  C3_WQ, C3_BQ, C3_WK, C3_BK, C3_WV, C3_BV, C3_WE, C3_BE, C3_WSKIP, C3_BSKIP, C3_WBETA,
  BN3_G, BN3_B, BN3_RM, BN3_RV,
  JK_WIH_F, JK_WHH_F, JK_BIH_F, JK_BHH_F, JK_WIH_B, JK_WHH_B, JK_BIH_B, JK_BHH_B, JK_ATT_W, JK_ATT_B,
  P_PROJ_W, P_PROJ_B, P_CLS_W1, P_CLS_B1, P_CLS_W2, P_CLS_B2, P_CLS_W3, P_CLS_B3
};

extern "C" void kernel_launch(void* const* d_in, const int* in_sizes, int n_in,
                              void* d_out, int out_size, void* d_ws, size_t ws_size,
                              hipStream_t stream) {
  (void)n_in; (void)out_size; (void)ws_size;
  const int NN = in_sizes[IN_X] / 64;     // 20000
  const int E  = in_sizes[IN_EA] / 3;     // 320000

  const float* x     = (const float*)d_in[IN_X];
  const int*   ei    = (const int*)d_in[IN_EI];
  const int*   src   = ei;
  const int*   dst   = ei + E;
  const float* eattr = (const float*)d_in[IN_EA];
  auto P = [&](int i){ return (const float*)d_in[i]; };

  // ---------------- workspace partition ----------------
  size_t off = 0;
  auto alloc = [&](size_t bytes) -> char* {
    char* p = (char*)d_ws + off;
    off += (bytes + 255) & ~(size_t)255;
    return p;
  };
  float*     x0    = (float*)alloc((size_t)NN * 256 * 4);
  float*     ebuf  = (float*)alloc((size_t)E * EDIM * 4);
  _Float16*  e16   = (_Float16*)alloc((size_t)E * EDIM * 2);
  _Float16*  ea16  = (_Float16*)alloc((size_t)E * DD * 2);
  float*     xl    = (float*)alloc((size_t)NN * DD * 4);
  float*     xr    = (float*)alloc((size_t)NN * DD * 4);
  float*     vbuf  = (float*)alloc((size_t)NN * DD * 4);
  float*     xskip = (float*)alloc((size_t)NN * DD * 4);
  float*     aggr  = (float*)alloc((size_t)NN * DD * 4);
  float*     l1    = (float*)alloc((size_t)NN * DD * 4);
  float*     l2    = (float*)alloc((size_t)NN * DD * 4);
  float*     l3    = (float*)alloc((size_t)NN * DD * 4);
  float*     logits= (float*)alloc((size_t)E * NHEADS * 4);
  float*     segmax= (float*)alloc((size_t)NN * NHEADS * 4);
  float*     segsum= (float*)alloc((size_t)NN * NHEADS * 4);
  _Float16*  A16   = (_Float16*)alloc((size_t)NN * DD * 2);
  _Float16*  h16   = (_Float16*)alloc((size_t)NN * JKH * 2);
  _Float16*  B16   = (_Float16*)alloc((size_t)DD * DD * 2);
  _Float16*  wihF  = (_Float16*)alloc((size_t)DD * G4 * 2);
  _Float16*  whhF  = (_Float16*)alloc((size_t)JKH * G4 * 2);
  _Float16*  wihB  = (_Float16*)alloc((size_t)DD * G4 * 2);
  _Float16*  whhB  = (_Float16*)alloc((size_t)JKH * G4 * 2);
  float*     G     = (float*)alloc((size_t)NN * G4 * 4);
  float*     hcur  = (float*)alloc((size_t)NN * JKH * 4);
  float*     ccur  = (float*)alloc((size_t)NN * JKH * 4);
  float*     hf    = (float*)alloc((size_t)3 * NN * JKH * 4);
  float*     hb    = (float*)alloc((size_t)3 * NN * JKH * 4);
  float*     jk    = (float*)alloc((size_t)NN * DD * 4);
  float*     emb   = (float*)alloc((size_t)NN * CCH * 4);
  float*     h1    = (float*)alloc((size_t)NN * 64 * 4);
  float*     h2    = (float*)alloc((size_t)NN * 32 * 4);

  // ---------------- launch helpers ----------------
  auto fill = [&](float* p, float v, size_t n){
    k_fill<<<(unsigned)((n + 255) / 256), 256, 0, stream>>>(p, v, n);
  };
  auto cvt = [&](const float* s, _Float16* d, size_t n){
    k_cvt16<<<(unsigned)((n + 255) / 256), 256, 0, stream>>>(s, d, n);
  };
  auto cvtT = [&](const float* s, _Float16* d, int R, int C){
    size_t n = (size_t)R * C;
    k_cvt16_tr<<<(unsigned)((n + 255) / 256), 256, 0, stream>>>(s, d, R, C);
  };
  auto gemm = [&](const _Float16* A, const _Float16* B, const float* bias, float* C,
                  int M, int Nn, int K, int flags){
    dim3 g((M + 127) / 128, (Nn + 63) / 64);
    k_gemm_wmma<<<g, 256, 0, stream>>>(A, B, bias, C, M, Nn, K, flags);
  };
  const int nwE = E * NHEADS;

  // ---------------- 1. input projection + LN + ELU ----------------
  cvt(x, A16, (size_t)NN * 64);
  cvt(P(P_IN_W), B16, (size_t)64 * 256);
  gemm(A16, B16, P(P_IN_B), x0, NN, 256, 64, 1);
  k_ln_elu<<<NN, 256, 0, stream>>>(x0, P(P_LN_G), P(P_LN_B));

  // ---------------- 2. edge embedding ----------------
  k_edge_embed<<<(E + 255) / 256, 256, 0, stream>>>(
      eattr, P(P_EM_W1), P(P_EM_B1), P(P_EM_W2), P(P_EM_B2), ebuf, E);
  cvt(ebuf, e16, (size_t)E * EDIM);

  // ---------------- GATv2 layers ----------------
  auto gat_layer = [&](const float* xin, int fin,
                       const float* Wl, const float* bl, const float* Wr, const float* br,
                       const float* We, const float* att, const float* cbias,
                       const float* bg, const float* bb, const float* brm, const float* brv,
                       const float* res, float* lout){
    cvt(xin, A16, (size_t)NN * fin);
    cvt(Wl, B16, (size_t)fin * DD);
    gemm(A16, B16, bl, xl, NN, DD, fin, 1);
    cvt(Wr, B16, (size_t)fin * DD);
    gemm(A16, B16, br, xr, NN, DD, fin, 1);
    cvt(We, B16, (size_t)EDIM * DD);
    gemm(e16, B16, nullptr, (float*)ea16, E, DD, EDIM, 8);   // f16 out
    fill(segmax, -3.0e38f, (size_t)NN * NHEADS);
    fill(segsum, 0.f, (size_t)NN * NHEADS);
    fill(aggr, 0.f, (size_t)NN * DD);
    k_gat_logits<<<(nwE + 7) / 8, 256, 0, stream>>>(xl, xr, ea16, src, dst, att, logits, segmax, E);
    k_seg_expsum<<<(nwE + 255) / 256, 256, 0, stream>>>(logits, dst, segmax, segsum, E);
    k_gat_scatter<<<(nwE + 7) / 8, 256, 0, stream>>>(xl, logits, segsum, src, dst, aggr, E);
    k_bn_elu<<<(unsigned)(((size_t)NN * DD + 255) / 256), 256, 0, stream>>>(
        aggr, cbias, bg, bb, brm, brv, res, lout, (size_t)NN * DD);
  };
  gat_layer(x0, 256, P(C1_WL), P(C1_BL), P(C1_WR), P(C1_BR), P(C1_WE), P(C1_ATT), P(C1_BIAS),
            P(BN1_G), P(BN1_B), P(BN1_RM), P(BN1_RV), nullptr, l1);
  gat_layer(l1, 512, P(C2_WL), P(C2_BL), P(C2_WR), P(C2_BR), P(C2_WE), P(C2_ATT), P(C2_BIAS),
            P(BN2_G), P(BN2_B), P(BN2_RM), P(BN2_RV), l1, l2);

  // ---------------- TransformerConv layer ----------------
  cvt(l2, A16, (size_t)NN * DD);
  cvt(P(C3_WQ), B16, (size_t)DD * DD);  gemm(A16, B16, P(C3_BQ), xl, NN, DD, DD, 1);   // q
  cvt(P(C3_WK), B16, (size_t)DD * DD);  gemm(A16, B16, P(C3_BK), xr, NN, DD, DD, 1);   // k
  cvt(P(C3_WV), B16, (size_t)DD * DD);  gemm(A16, B16, P(C3_BV), vbuf, NN, DD, DD, 1); // v
  cvt(P(C3_WE), B16, (size_t)EDIM * DD);
  gemm(e16, B16, P(C3_BE), (float*)ea16, E, DD, EDIM, 1 | 8);                          // ea (f16)
  fill(segmax, -3.0e38f, (size_t)NN * NHEADS);
  fill(segsum, 0.f, (size_t)NN * NHEADS);
  fill(aggr, 0.f, (size_t)NN * DD);
  k_trans_logits<<<(nwE + 7) / 8, 256, 0, stream>>>(xl, xr, ea16, src, dst, logits, segmax, E);
  k_seg_expsum<<<(nwE + 255) / 256, 256, 0, stream>>>(logits, dst, segmax, segsum, E);
  k_trans_scatter<<<(nwE + 7) / 8, 256, 0, stream>>>(vbuf, ea16, logits, segsum, src, dst, aggr, E);
  cvt(P(C3_WSKIP), B16, (size_t)DD * DD);
  gemm(A16, B16, P(C3_BSKIP), xskip, NN, DD, DD, 1);                                   // A16 still l2
  k_beta<<<(NN + 7) / 8, 256, 0, stream>>>(aggr, xskip, P(C3_WBETA), vbuf, NN);
  k_bn_elu<<<(unsigned)(((size_t)NN * DD + 255) / 256), 256, 0, stream>>>(
      vbuf, nullptr, P(BN3_G), P(BN3_B), P(BN3_RM), P(BN3_RV), nullptr, l3, (size_t)NN * DD);

  // ---------------- JK bidirectional LSTM ----------------
  cvtT(P(JK_WIH_F), wihF, G4, DD);
  cvtT(P(JK_WHH_F), whhF, G4, JKH);
  cvtT(P(JK_WIH_B), wihB, G4, DD);
  cvtT(P(JK_WHH_B), whhB, G4, JKH);
  const float* xs[3] = { l1, l2, l3 };
  auto lstm_dir = [&](const _Float16* wih, const _Float16* whh,
                      const float* bih, const float* bhh, float* hs, bool fwd){
    fill(hcur, 0.f, (size_t)NN * JKH);
    fill(ccur, 0.f, (size_t)NN * JKH);
    for (int t = 0; t < 3; t++){
      int l = fwd ? t : (2 - t);
      cvt(xs[l], A16, (size_t)NN * DD);
      gemm(A16, wih, nullptr, G, NN, G4, DD, 0);
      cvt(hcur, h16, (size_t)NN * JKH);
      gemm(h16, whh, nullptr, G, NN, G4, JKH, 4);   // accumulate
      k_lstm_cell<<<(unsigned)(((size_t)NN * JKH + 255) / 256), 256, 0, stream>>>(
          G, bih, bhh, hcur, ccur, hs + (size_t)l * NN * JKH, NN);
    }
  };
  lstm_dir(wihF, whhF, P(JK_BIH_F), P(JK_BHH_F), hf, true);
  lstm_dir(wihB, whhB, P(JK_BIH_B), P(JK_BHH_B), hb, false);
  k_jk_attn<<<(NN + 7) / 8, 256, 0, stream>>>(hf, hb, P(JK_ATT_W), P(JK_ATT_B), l1, l2, l3, jk, NN);

  // ---------------- projection + classifier ----------------
  cvt(jk, A16, (size_t)NN * DD);
  cvt(P(P_PROJ_W), B16, (size_t)DD * CCH);
  gemm(A16, B16, P(P_PROJ_B), emb, NN, CCH, DD, 1);
  cvt(emb, A16, (size_t)NN * CCH);
  cvt(P(P_CLS_W1), B16, (size_t)CCH * 64);
  gemm(A16, B16, P(P_CLS_B1), h1, NN, 64, CCH, 3);         // bias + ELU
  cvt(h1, A16, (size_t)NN * 64);
  cvt(P(P_CLS_W2), B16, (size_t)64 * 32);
  gemm(A16, B16, P(P_CLS_B2), h2, NN, 32, 64, 3);          // bias + ELU
  cvt(h2, A16, (size_t)NN * 32);
  cvt(P(P_CLS_W3), B16, (size_t)32 * 2);
  gemm(A16, B16, P(P_CLS_B3), (float*)d_out, NN, 2, 32, 1); // logits [N,2]
}
